// conv_lstm_1331439862032
// MI455X (gfx1250) — matrix operand order repaired and checked by reading, not compiled, not run on hardware
//
#include <hip/hip_runtime.h>
#include <stdint.h>

typedef __attribute__((ext_vector_type(16))) __bf16 v16bf;
typedef __attribute__((ext_vector_type(8)))  __bf16 v8bf;
typedef __attribute__((ext_vector_type(8)))  float  v8f;
typedef __attribute__((ext_vector_type(4)))  int    v4i;

#define B_   32
#define T_   1024
#define CIN  128
#define F_   256
#define KW   5
#define TP   510            // (1024-5)/2 + 1
#define H_   512
#define G4   2048
#define MCNV (TP*B_)        // 16320
#define KC   (CIN*KW)       // 640
#define NWG  32             // persistent LSTM workgroups

#define SBK  128            // K-superblock staged in LDS
#define LDSP (SBK + 8)      // +16B pad -> row stride 272B, kills 64-bank row conflicts

#if __has_builtin(__builtin_amdgcn_global_load_async_to_lds_b128) && __has_builtin(__builtin_amdgcn_s_wait_asynccnt)
#define HAS_ASYNC_LDS 1
#else
#define HAS_ASYNC_LDS 0
#endif

// ---------------- helpers ----------------

__device__ __forceinline__ unsigned short f2bf(float f){
  unsigned u = __float_as_uint(f);
  unsigned r = (u + 0x7FFFu + ((u >> 16) & 1u)) >> 16;   // RNE
  return (unsigned short)r;
}
__device__ __forceinline__ float sigmoidf(float x){ return 1.0f/(1.0f + __expf(-x)); }

// 16-bit A-matrix 16x32 fragment (ISA 7.12.2): lane<16 holds K[kb..kb+7] and K[kb+16..kb+23], kb=(lane/16)*8
__device__ __forceinline__ v16bf load_a_frag(const unsigned short* rowptr, int lane){
  int kb = (lane >> 4) * 8;
  union { v16bf v; v8bf h[2]; } u;
  u.h[0] = *(const v8bf*)(rowptr + kb);
  u.h[1] = *(const v8bf*)(rowptr + kb + 16);
  return u.v;
}
// 16-bit B-matrix 32x16 fragment: lane = column N (lane&15), K[kb..kb+7] and K[kb+16..kb+23], kb=(lane/16)*8
__device__ __forceinline__ v16bf load_b_frag(const unsigned short* nrowptr, int lane){
  int kb = (lane >> 4) * 8;
  union { v16bf v; v8bf h[2]; } u;
  u.h[0] = *(const v8bf*)(nrowptr + kb);
  u.h[1] = *(const v8bf*)(nrowptr + kb + 16);
  return u.v;
}
__device__ __forceinline__ v8f wmma_bf16(v16bf a, v16bf b, v8f c){
  return __builtin_amdgcn_wmma_f32_16x16x32_bf16(false, a, false, b, (short)0, c, false, false);
}

// copy one 16B chunk global -> LDS (ASYNCcnt path if available)
__device__ __forceinline__ void stage16(unsigned short* lds_dst, const unsigned short* g_src){
#if HAS_ASYNC_LDS
  __builtin_amdgcn_global_load_async_to_lds_b128(
      (__attribute__((address_space(1))) v4i*)g_src,
      (__attribute__((address_space(3))) v4i*)lds_dst, 0, 0);
#else
  *(v8bf*)lds_dst = *(const v8bf*)g_src;
#endif
}
__device__ __forceinline__ void stage_wait(){
#if HAS_ASYNC_LDS
  __builtin_amdgcn_s_wait_asynccnt(0);
#endif
}

// ---------------- prep kernels ----------------

__global__ void k_cast_bf16(const float* __restrict__ src, unsigned short* __restrict__ dst, int n){
  int i = blockIdx.x * 256 + threadIdx.x;
  if (i < n) dst[i] = f2bf(src[i]);
}

// wc[f*KC + k*CIN + c] = conv_w[f][c][k]  (B-matrix [N=F][K=KC])
__global__ void k_prep_convw(const float* __restrict__ cw, unsigned short* __restrict__ wc){
  int i = blockIdx.x * 256 + threadIdx.x;
  if (i < F_*KC){
    int f = i / KC, r = i % KC, k = r / CIN, c = r % CIN;
    wc[i] = f2bf(cw[(f*CIN + c)*KW + k]);
  }
}

__global__ void k_prep_misc(const float* __restrict__ gam, const float* __restrict__ bet,
                            const float* __restrict__ mean, const float* __restrict__ var,
                            const float* __restrict__ bih, const float* __restrict__ bhh,
                            float* __restrict__ bnscale, float* __restrict__ bnshift,
                            float* __restrict__ bias){
  int i = blockIdx.x * 256 + threadIdx.x;
  if (i < F_){
    float inv = gam[i] * rsqrtf(var[i] + 1e-5f);
    bnscale[i] = inv;
    bnshift[i] = bet[i] - mean[i]*inv;
  }
  if (i < G4) bias[i] = bih[i] + bhh[i];
}

__global__ void k_init(unsigned short* __restrict__ hbf, float* __restrict__ hf,
                       float* __restrict__ cf, float* __restrict__ uf, unsigned* __restrict__ cnt){
  int i = blockIdx.x * 256 + threadIdx.x;
  if (i < B_*H_){ hbf[i] = 0; hf[i] = 0.f; cf[i] = 0.f; }
  if (i < B_) uf[i] = 1.f;
  if (i == 0) *cnt = 0u;
}

// ---------------- conv + relu + bn (GEMM, M=16320 K=640 N=256) ----------------
// grid(128,4), block 256 (8 waves -> 128M x 64N tile).
// B panel staged in double-buffered LDS; stage(s+1) overlaps compute(s) via ASYNCcnt.
__global__ void __launch_bounds__(256, 1)
k_conv(const unsigned short* __restrict__ xb,   // [B,T,CIN] bf16
       const unsigned short* __restrict__ wc,   // [F,KC]    bf16
       const float* __restrict__ convb,
       const float* __restrict__ bnscale, const float* __restrict__ bnshift,
       unsigned short* __restrict__ y){
  __shared__ unsigned short lsb[2][64 * LDSP];
  int tid   = threadIdx.x;
  int lane  = tid & 31;
  int wave  = tid >> 5;
  int mbase = blockIdx.x * 128 + wave * 16;
  int nbase = blockIdx.y * 64;
  int l15   = lane & 15;

  int arow = mbase + l15;
  if (arow >= MCNV) arow = MCNV - 1;          // clamp loads; stores predicated
  int t = arow / B_, bb = arow % B_;
  const unsigned short* xrow = xb + (size_t)bb * T_ * CIN;

  // this thread's 4 staging chunks (fixed row/col, varying superblock)
  int srow = (tid * 4) >> 4;                   // 0..63
  int scol0 = ((tid * 4) & 15) * 8;            // 0,32,64,96 pattern start
  const unsigned short* wrow = wc + (size_t)(nbase + srow) * KC;

  v8f acc[4] = {v8f{}, v8f{}, v8f{}, v8f{}};
  const int NSB = KC / SBK;                    // 5

  // prologue: stage superblock 0 into buffer 0
  #pragma unroll
  for (int c = 0; c < 4; ++c)
    stage16(&lsb[0][srow * LDSP + scol0 + c*8], wrow + scol0 + c*8);

  for (int sb = 0; sb < NSB; ++sb){
    int buf = sb & 1;
    stage_wait();
    __syncthreads();                           // buffer `buf` published

    if (sb + 1 < NSB){                         // overlap: stage next superblock
      #pragma unroll
      for (int c = 0; c < 4; ++c)
        stage16(&lsb[buf ^ 1][srow * LDSP + scol0 + c*8],
                wrow + (sb + 1) * SBK + scol0 + c*8);
    }

    // A fragments for this superblock (clause-batched)
    int kc0 = sb * SBK;
    v16bf afr[4];
    #pragma unroll
    for (int kk = 0; kk < 4; ++kk){
      int kc = kc0 + kk * 32;
      int k = kc >> 7, c0 = kc & 127;          // chunks never straddle a tap (128%32==0)
      afr[kk] = load_a_frag(xrow + (size_t)(2*t + k) * CIN + c0, lane);
    }

    // register-double-buffered B fragments from LDS
    const unsigned short* lb = &lsb[buf][l15 * LDSP];
    v16bf bcur[4];
    #pragma unroll
    for (int i = 0; i < 4; ++i) bcur[i] = load_b_frag(lb + i*16*LDSP, lane);
    #pragma unroll
    for (int kk = 0; kk < 4; ++kk){
      v16bf bnx[4];
      int kk2 = (kk < 3) ? kk + 1 : kk;
      #pragma unroll
      for (int i = 0; i < 4; ++i) bnx[i] = load_b_frag(lb + i*16*LDSP + kk2*32, lane);
      #pragma unroll
      for (int i = 0; i < 4; ++i) acc[i] = wmma_bf16(afr[kk], bcur[i], acc[i]);
      #pragma unroll
      for (int i = 0; i < 4; ++i) bcur[i] = bnx[i];
    }
    __syncthreads();                           // done reading `buf`
  }

  int mm0 = (lane >> 4) * 8;
  #pragma unroll
  for (int i = 0; i < 4; ++i){
    int n = nbase + i*16 + l15;
    float cb = convb[n], sc = bnscale[n], sh = bnshift[n];
    #pragma unroll
    for (int j = 0; j < 8; ++j){
      int m = mbase + mm0 + j;
      if (m < MCNV){
        float v = acc[i][j] + cb;
        v = v > 0.f ? v : 0.f;
        y[(size_t)m * F_ + n] = f2bf(v * sc + sh);
      }
    }
  }
}

// ---------------- x_proj GEMM (M=16320 K=256 N=2048) ----------------
// grid(128,32), block 256. Same double-buffered staging, 2 superblocks.
__global__ void __launch_bounds__(256, 1)
k_xproj(const unsigned short* __restrict__ y,    // [MCNV,F] bf16
        const unsigned short* __restrict__ wih,  // [G4,F]   bf16
        const float* __restrict__ bias,          // [G4]
        float* __restrict__ xp){
  __shared__ unsigned short lsb[2][64 * LDSP];
  int tid   = threadIdx.x;
  int lane  = tid & 31;
  int wave  = tid >> 5;
  int mbase = blockIdx.x * 128 + wave * 16;
  int nbase = blockIdx.y * 64;
  int l15   = lane & 15;

  int arow = mbase + l15;
  if (arow >= MCNV) arow = MCNV - 1;
  const unsigned short* ap0 = y + (size_t)arow * F_;

  int srow = (tid * 4) >> 4;
  int scol0 = ((tid * 4) & 15) * 8;
  const unsigned short* wrow = wih + (size_t)(nbase + srow) * F_;

  v8f acc[4] = {v8f{}, v8f{}, v8f{}, v8f{}};
  const int NSB = F_ / SBK;                    // 2

  #pragma unroll
  for (int c = 0; c < 4; ++c)
    stage16(&lsb[0][srow * LDSP + scol0 + c*8], wrow + scol0 + c*8);

  for (int sb = 0; sb < NSB; ++sb){
    int buf = sb & 1;
    stage_wait();
    __syncthreads();

    if (sb + 1 < NSB){
      #pragma unroll
      for (int c = 0; c < 4; ++c)
        stage16(&lsb[buf ^ 1][srow * LDSP + scol0 + c*8],
                wrow + (sb + 1) * SBK + scol0 + c*8);
    }

    int kc0 = sb * SBK;
    v16bf afr[4];
    #pragma unroll
    for (int kk = 0; kk < 4; ++kk)
      afr[kk] = load_a_frag(ap0 + kc0 + kk*32, lane);

    const unsigned short* lb = &lsb[buf][l15 * LDSP];
    v16bf bcur[4];
    #pragma unroll
    for (int i = 0; i < 4; ++i) bcur[i] = load_b_frag(lb + i*16*LDSP, lane);
    #pragma unroll
    for (int kk = 0; kk < 4; ++kk){
      v16bf bnx[4];
      int kk2 = (kk < 3) ? kk + 1 : kk;
      #pragma unroll
      for (int i = 0; i < 4; ++i) bnx[i] = load_b_frag(lb + i*16*LDSP + kk2*32, lane);
      #pragma unroll
      for (int i = 0; i < 4; ++i) acc[i] = wmma_bf16(afr[kk], bcur[i], acc[i]);
      #pragma unroll
      for (int i = 0; i < 4; ++i) bcur[i] = bnx[i];
    }
    __syncthreads();
  }

  int mm0 = (lane >> 4) * 8;
  #pragma unroll
  for (int i = 0; i < 4; ++i){
    int n = nbase + i*16 + l15;
    float bs = bias[n];
    #pragma unroll
    for (int j = 0; j < 8; ++j){
      int m = mbase + mm0 + j;
      if (m < MCNV) xp[(size_t)m * G4 + n] = acc[i][j] + bs;
    }
  }
}

// ---------------- persistent skip-LSTM ----------------

__device__ __forceinline__ void gridsync(unsigned* cnt, unsigned* round){
  __threadfence();
  __syncthreads();
  if (threadIdx.x == 0){
    *round += 1;
    unsigned target = (*round) * NWG;
    __hip_atomic_fetch_add(cnt, 1u, __ATOMIC_ACQ_REL, __HIP_MEMORY_SCOPE_AGENT);
    while (__hip_atomic_load(cnt, __ATOMIC_ACQUIRE, __HIP_MEMORY_SCOPE_AGENT) < target)
      __builtin_amdgcn_s_sleep(1);
  }
  __syncthreads();
}

// grid(NWG=32) x 256 threads. 256 waves == 2 M-tiles x 128 N-tiles per step.
// Each wave's 16x512 slice of w_hh is loop-invariant -> held in 128 VGPRs for all 510 steps.
__global__ void __launch_bounds__(256, 1)
k_lstm(const float* __restrict__ xp,            // [MCNV,G4]
       const unsigned short* __restrict__ whh,  // [G4,H] bf16
       const float* __restrict__ wuh,           // [H]
       const float* __restrict__ buh,           // [1]
       unsigned short* __restrict__ hbf,        // [B,H] bf16
       float* __restrict__ hf, float* __restrict__ cf,
       float* __restrict__ uf,                  // [B]
       float* __restrict__ gates,               // [B,G4]
       unsigned* __restrict__ barcnt){
  __shared__ float red[256];
  int lane = threadIdx.x & 31;
  int wave = threadIdx.x >> 5;
  int W  = blockIdx.x * 8 + wave;      // 0..255
  int m0 = (W >> 7) * 16;              // batch tile: rows 0-15 or 16-31
  int n0 = (W & 127) * 16;             // gate columns
  unsigned round = 0;

  const unsigned short* arow = hbf + (size_t)(m0 + (lane & 15)) * H_;
  const unsigned short* brow = whh + (size_t)(n0 + (lane & 15)) * H_;
  int b = blockIdx.x;                  // elementwise phase: WG b owns batch row b

  v16bf bfr[16];
  #pragma unroll
  for (int kk = 0; kk < 16; ++kk) bfr[kk] = load_b_frag(brow + kk*32, lane);

  int n   = n0 + (lane & 15);
  int mm0 = (lane >> 4) * 8;

  for (int t = 0; t < TP; ++t){
    const float* xpr = xp + (size_t)t * B_ * G4;
    #pragma unroll
    for (int j = 0; j < 8; ++j)
      __builtin_prefetch(&xpr[(size_t)(m0 + mm0 + j) * G4 + n], 0, 1);

    // ---- gates = x_proj[t] + h @ w_hh^T ----
    v16bf afr[16];
    #pragma unroll
    for (int kk = 0; kk < 16; ++kk) afr[kk] = load_a_frag(arow + kk*32, lane);
    v8f acc = {};
    #pragma unroll
    for (int kk = 0; kk < 16; ++kk) acc = wmma_bf16(afr[kk], bfr[kk], acc);

    #pragma unroll
    for (int j = 0; j < 8; ++j){
      int m = m0 + mm0 + j;
      gates[(size_t)m * G4 + n] = acc[j] + xpr[(size_t)m * G4 + n];
    }
    gridsync(barcnt, &round);

    // ---- elementwise skip-LSTM update for batch row b ----
    float uo = uf[b];
    float ub = rintf(uo);              // straight-through forward = round(u), RNE like jnp.round
    float part = 0.f;
    for (int j = threadIdx.x; j < H_; j += 256){
      const float* gr = gates + (size_t)b * G4;
      float gi = gr[j], gfv = gr[512 + j], gg = gr[1024 + j], go = gr[1536 + j];
      float co = cf[b*H_ + j], ho = hf[b*H_ + j];
      float ct = sigmoidf(gfv) * co + sigmoidf(gi) * tanhf(gg);
      float ht = sigmoidf(go) * tanhf(ct);
      float cn = ub*ct + (1.f-ub)*co;
      float hn = ub*ht + (1.f-ub)*ho;
      cf[b*H_ + j] = cn;
      hf[b*H_ + j] = hn;
      hbf[b*H_ + j] = f2bf(hn);
      part += cn * wuh[j];
    }
    red[threadIdx.x] = part;
    __syncthreads();
    #pragma unroll
    for (int s = 128; s > 0; s >>= 1){
      if (threadIdx.x < s) red[threadIdx.x] += red[threadIdx.x + s];
      __syncthreads();
    }
    if (threadIdx.x == 0){
      float du = sigmoidf(red[0] + buh[0]);
      uf[b] = ub*du + (1.f-ub)*(uo + fminf(du, 1.f - uo));
    }
    gridsync(barcnt, &round);
  }
}

__global__ void k_copyout(const float* __restrict__ hf, float* __restrict__ out){
  int i = blockIdx.x * 256 + threadIdx.x;
  if (i < B_*H_) out[i] = hf[i];
}

// ---------------- host ----------------

extern "C" void kernel_launch(void* const* d_in, const int* in_sizes, int n_in,
                              void* d_out, int out_size, void* d_ws, size_t ws_size,
                              hipStream_t stream){
  const float* x      = (const float*)d_in[0];
  const float* conv_w = (const float*)d_in[1];
  const float* conv_b = (const float*)d_in[2];
  const float* gam    = (const float*)d_in[3];
  const float* bet    = (const float*)d_in[4];
  const float* mean   = (const float*)d_in[5];
  const float* var    = (const float*)d_in[6];
  const float* w_ih   = (const float*)d_in[7];
  const float* w_hh   = (const float*)d_in[8];
  const float* b_ih   = (const float*)d_in[9];
  const float* b_hh   = (const float*)d_in[10];
  const float* w_uh   = (const float*)d_in[11];
  const float* b_uh   = (const float*)d_in[12];
  float* out = (float*)d_out;

  char* ws = (char*)d_ws;
  size_t off = 0;
  auto take = [&](size_t bytes)->char*{ char* p = ws + off; off = (off + bytes + 255) & ~(size_t)255; return p; };

  unsigned short* xb   = (unsigned short*)take((size_t)B_*T_*CIN*2);
  unsigned short* wc   = (unsigned short*)take((size_t)F_*KC*2);
  unsigned short* wih  = (unsigned short*)take((size_t)G4*F_*2);
  unsigned short* whh  = (unsigned short*)take((size_t)G4*H_*2);
  unsigned short* y    = (unsigned short*)take((size_t)MCNV*F_*2);
  float* xp            = (float*)take((size_t)MCNV*G4*4);
  float* bias          = (float*)take((size_t)G4*4);
  float* bnscale       = (float*)take((size_t)F_*4);
  float* bnshift       = (float*)take((size_t)F_*4);
  float* gates         = (float*)take((size_t)B_*G4*4);
  unsigned short* hbf  = (unsigned short*)take((size_t)B_*H_*2);
  float* hf            = (float*)take((size_t)B_*H_*4);
  float* cf            = (float*)take((size_t)B_*H_*4);
  float* uf            = (float*)take((size_t)B_*4);
  unsigned* barcnt     = (unsigned*)take(256);

  // prep
  int nx = B_*T_*CIN;
  k_cast_bf16<<<(nx+255)/256, 256, 0, stream>>>(x, xb, nx);
  k_cast_bf16<<<(G4*F_+255)/256, 256, 0, stream>>>(w_ih, wih, G4*F_);
  k_cast_bf16<<<(G4*H_+255)/256, 256, 0, stream>>>(w_hh, whh, G4*H_);
  k_prep_convw<<<(F_*KC+255)/256, 256, 0, stream>>>(conv_w, wc);
  k_prep_misc<<<(G4+255)/256, 256, 0, stream>>>(gam, bet, mean, var, b_ih, b_hh, bnscale, bnshift, bias);
  k_init<<<(B_*H_+255)/256, 256, 0, stream>>>(hbf, hf, cf, uf, barcnt);

  // conv + relu + bn
  k_conv<<<dim3(128, 4), 256, 0, stream>>>(xb, wc, conv_b, bnscale, bnshift, y);
  // x_proj
  k_xproj<<<dim3(128, 32), 256, 0, stream>>>(y, wih, bias, xp);
  // persistent skip-LSTM scan
  k_lstm<<<NWG, 256, 0, stream>>>(xp, whh, w_uh, b_uh, hbf, hf, cf, uf, gates, barcnt);
  // final h -> out
  k_copyout<<<(B_*H_+255)/256, 256, 0, stream>>>(hf, out);
}
